// BowEncoder_35373350650620
// MI455X (gfx1250) — matrix operand order repaired and checked by reading, not compiled, not run on hardware
//
#include <hip/hip_runtime.h>
#include <hip/hip_bf16.h>

// BoW mean-pooling encoder for MI455X (gfx1250, wave32).
// out[b, h] = (1/len_b) * sum_{t < len_b} emb[ids[b,t], h]
//
// Memory-bound: up to 128 MB of gathered f32 emb rows -> ~5.5 us at 23.3 TB/s.
// Reduction over T runs on v_wmma_f32_16x16x32_f16: A = 1.0/0.0 validity
// weights in row M=0 (f16-exact), B = gathered emb values (f32->f16 cvt),
// C/D accumulate in f32; 1/len applied once in f32 at the end.
//
// Round-2 changes: 32-bit element offsets against the uniform emb base
// (GVS addressing + scale_offset, H=256 specialization folds id*H to a
// shift) and a 2x T-unroll for a wider outstanding-load window.

typedef __attribute__((ext_vector_type(16))) _Float16 v16h;
typedef __attribute__((ext_vector_type(8)))  float    v8f;

template<int HC>   // HC = compile-time hidden size (0 => use runtime h)
__global__ __launch_bounds__(32)
void bow_wmma_kernel(const int* __restrict__ ids,    // [B, T]
                     const int* __restrict__ lens,   // [B]
                     const float* __restrict__ emb,  // [V, H]
                     float* __restrict__ out,        // [B, H]
                     int T, int h)
{
    const int H = HC ? HC : h;

    const int hblock = blockIdx.x;            // 16 hidden columns per wave
    const int b      = blockIdx.y;
    const int lane   = threadIdx.x;           // wave32
    const int n      = lane & 15;             // B/D column index
    const int kbase  = (lane & 16) ? 16 : 0;  // B-layout: upper lanes hold K=16..31
    const bool m0    = ((lane & 15) == 0);    // A row M=0 lives in lanes 0 and 16

    const int      len   = lens[b];
    const int      hbase = hblock * 16;
    const unsigned hoff  = (unsigned)(hbase + n);
    const int*     idrow = ids + b * T;

    // Gather one 32x16 B tile: slot s = emb[id(t0 + kbase + s)][hbase + n].
    // 32-bit element offset vs the uniform emb base -> GVS + scale_offset.
    auto gather = [&](int idv) -> v16h {
        v16h bm;
#pragma unroll
        for (int s = 0; s < 16; ++s) {
            int id = __shfl(idv, kbase + s, 32);
            bm[s]  = (_Float16)emb[(unsigned)id * (unsigned)H + hoff];
        }
        return bm;
    };

    // A-matrix (16x32 f16), all-valid: row 0 = 1.0, rest 0.
    v16h am_full;
#pragma unroll
    for (int s = 0; s < 16; ++s)
        am_full[s] = m0 ? (_Float16)1.0f : (_Float16)0.0f;

    v8f c = {};                               // 16x16 f32 accumulator

    const int nfull = len >> 5;               // full 32-timestep chunks
    const int rem   = len & 31;

    int t0 = 0;
    // 2x-unrolled main loop: 64 timesteps, two B tiles, two chained WMMAs.
    for (int it = 0; it + 1 < nfull; it += 2, t0 += 64) {
        int idv0 = idrow[t0 + lane];
        int idv1 = idrow[t0 + 32 + lane];
        __builtin_prefetch(&idrow[t0 + 64 + lane], 0, 0);   // global_prefetch_b8

        v16h bm0 = gather(idv0);
        v16h bm1 = gather(idv1);

        c = __builtin_amdgcn_wmma_f32_16x16x32_f16(
                false, am_full, false, bm0, (short)0, c, false, false);
        c = __builtin_amdgcn_wmma_f32_16x16x32_f16(
                false, am_full, false, bm1, (short)0, c, false, false);
    }
    if (nfull & 1) {                          // leftover full chunk
        int idv = idrow[t0 + lane];
        v16h bm = gather(idv);
        c = __builtin_amdgcn_wmma_f32_16x16x32_f16(
                false, am_full, false, bm, (short)0, c, false, false);
        t0 += 32;
    }
    if (rem) {
        // t0 + lane < T always (len <= T), so the id load is in-bounds;
        // t >= len rows are zeroed through A's weights, not through B.
        int idv = idrow[t0 + lane];
        v16h bm = gather(idv);
        // Masked A: slot s of this lane maps to K = kbase + s (same
        // contiguous order as B); keep weight only where t0 + K < len (K < rem).
        v16h am;
#pragma unroll
        for (int s = 0; s < 16; ++s) {
            int K = kbase + s;
            am[s] = (m0 && (K < rem)) ? (_Float16)1.0f : (_Float16)0.0f;
        }
        c = __builtin_amdgcn_wmma_f32_16x16x32_f16(
                false, am, false, bm, (short)0, c, false, false);
    }

    // D row M=0 sits in VGPR0 (c[0]) on lanes 0..15, N = lane.
    if (lane < 16)
        out[b * H + hbase + lane] = c[0] * (1.0f / (float)len);
}

extern "C" void kernel_launch(void* const* d_in, const int* in_sizes, int n_in,
                              void* d_out, int out_size, void* d_ws, size_t ws_size,
                              hipStream_t stream) {
    const int*   ids  = (const int*)d_in[0];   // input_ids   [B, T]
    const int*   lens = (const int*)d_in[1];   // input_lens  [B]
    const float* emb  = (const float*)d_in[2]; // emb         [V, H]
    float*       out  = (float*)d_out;         // context     [B, H]

    const int B = in_sizes[1];          // 64
    const int T = in_sizes[0] / B;      // 2048
    const int H = out_size / B;         // 256

    dim3 grid(H / 16, B);               // one wave per (hidden block, batch)
    if (H == 256) {
        bow_wmma_kernel<256><<<grid, 32, 0, stream>>>(ids, lens, emb, out, T, H);
    } else {
        bow_wmma_kernel<0><<<grid, 32, 0, stream>>>(ids, lens, emb, out, T, H);
    }
}
